// TreeEncoder_27152783245640
// MI455X (gfx1250) — matrix operand order repaired and checked by reading, not compiled, not run on hardware
//
#include <hip/hip_runtime.h>
#include <hip/hip_bf16.h>

typedef __attribute__((ext_vector_type(16))) __bf16 v16bf;
typedef __attribute__((ext_vector_type(8)))  __bf16 v8bf;
typedef __attribute__((ext_vector_type(4)))  __bf16 v4bf;
typedef __attribute__((ext_vector_type(8)))  float  v8f;

#define FEAT 128

// ---------------- utility kernels ----------------

__global__ void fill_f32(float* p, float v, int n) {
    int i = blockIdx.x * blockDim.x + threadIdx.x;
    if (i < n) p[i] = v;
}

__global__ void deg_edges(const int* __restrict__ ei, float* __restrict__ deg, int E) {
    int e = blockIdx.x * blockDim.x + threadIdx.x;
    if (e < E) atomicAdd(&deg[ei[E + e]], 1.0f);
}

__global__ void rsqrt_k(float* __restrict__ d, int n) {
    int i = blockIdx.x * blockDim.x + threadIdx.x;
    if (i < n) d[i] = rsqrtf(d[i]);          // deg >= 1 always (self-loop)
}

__global__ void f32_to_bf16(const float* __restrict__ in, __bf16* __restrict__ out, int n) {
    int i = blockIdx.x * blockDim.x + threadIdx.x;
    if (i < n) out[i] = (__bf16)in[i];
}

__global__ void relu_to_bf16(const float* __restrict__ in, __bf16* __restrict__ out, int n) {
    int i = blockIdx.x * blockDim.x + threadIdx.x;
    if (i < n) out[i] = (__bf16)fmaxf(in[i], 0.0f);
}

// Pre-swizzle a 128x128 row-major f32 weight (W[k][c]) into the per-lane
// WMMA B-matrix (32x16 per tile) layout, bf16:
//   Wsw[((kc*8 + n)*32 + lane)*16 + j] = W[kc*32 + (lane>>4)*16 + j][n*16 + (lane&15)]
// so each lane reads its 16-element fragment as one contiguous 32B load.
__global__ void swizzle_w(const float* __restrict__ W, __bf16* __restrict__ Wsw) {
    int idx = blockIdx.x * blockDim.x + threadIdx.x;
    if (idx >= 4 * 8 * 32 * 16) return;
    int j    = idx & 15;
    int lane = (idx >> 4) & 31;
    int n    = (idx >> 9) & 7;
    int kc   = idx >> 12;
    int c = n * 16 + (lane & 15);
    int k = kc * 32 + (lane >> 4) * 16 + j;
    Wsw[idx] = (__bf16)W[k * FEAT + c];
}

// ---------------- WMMA GEMM: [rows x 128](bf16) @ [128 x 128](swizzled bf16) ----------------
// One wave computes a 16-row strip x 128 cols. 32x v_wmma_f32_16x16x32_bf16 per strip.

__global__ void gemm128_wmma(const __bf16* __restrict__ A,
                             const __bf16* __restrict__ Bsw,
                             const float* __restrict__ bias,   // may be null
                             float* __restrict__ outf,         // used if outb == null
                             __bf16* __restrict__ outb,        // used if non-null
                             int rows)
{
    const int wave  = threadIdx.x >> 5;
    const int lane  = threadIdx.x & 31;
    const int strip = blockIdx.x * (blockDim.x >> 5) + wave;
    if (strip * 16 >= rows) return;            // wave-uniform: EXEC stays all-ones

    const int r  = lane & 15;                  // row (A) / col (B,C) within tile
    const int hi = lane >> 4;

    const __bf16* arow = A + (size_t)(strip * 16 + r) * FEAT;

    v8f acc[8] = {};

#pragma unroll
    for (int kc = 0; kc < 4; ++kc) {
        // A fragment: lanes 0-15 rows, contiguous K: hi half holds K 16..31 (matches swizzled B)
        v8bf alo = *(const v8bf*)(arow + kc * 32 + hi * 16);
        v8bf ahi = *(const v8bf*)(arow + kc * 32 + hi * 16 + 8);
        v16bf a = __builtin_shufflevector(alo, ahi,
                                          0, 1, 2, 3, 4, 5, 6, 7,
                                          8, 9, 10, 11, 12, 13, 14, 15);
#pragma unroll
        for (int n = 0; n < 8; ++n) {
            v16bf b = *(const v16bf*)(Bsw + (((kc * 8 + n) * 32 + lane) << 4));
            acc[n] = __builtin_amdgcn_wmma_f32_16x16x32_bf16(
                false, a, false, b, (short)0, acc[n], false, false);
        }
    }

    // C/D layout: lanes 0-15 -> N=lane, M=0..7 across v8f; lanes 16-31 -> M=8..15
#pragma unroll
    for (int n = 0; n < 8; ++n) {
        const int col = n * 16 + r;
        const float badd = bias ? bias[col] : 0.0f;
#pragma unroll
        for (int v = 0; v < 8; ++v) {
            const int row = strip * 16 + hi * 8 + v;
            float val = acc[n][v] + badd;
            if (outb) outb[(size_t)row * FEAT + col] = (__bf16)val;
            else      outf[(size_t)row * FEAT + col] = val;
        }
    }
}

// ---------------- GCN aggregation ----------------

// h[i][f] = t[i][f]*dis[i]^2 + bias[f]   (self-loop term; also initializes h)
__global__ void agg_init(const __bf16* __restrict__ t, const float* __restrict__ dis,
                         const float* __restrict__ bias, float* __restrict__ h, int n128)
{
    int gid = blockIdx.x * blockDim.x + threadIdx.x;
    if (gid >= n128) return;
    int i = gid >> 7;
    int f = gid & 127;
    float d = dis[i];
    h[gid] = (float)t[gid] * d * d + bias[f];
}

// h[dst] += t[src] * dis[src]*dis[dst]   — one wave (32 lanes x 4 feats) per edge
__global__ void agg_edges(const int* __restrict__ ei, const __bf16* __restrict__ t,
                          const float* __restrict__ dis, float* __restrict__ h, int E)
{
    int gid = blockIdx.x * blockDim.x + threadIdx.x;
    int e = gid >> 5;
    if (e >= E) return;
    int q   = gid & 31;
    int src = ei[e];
    int dst = ei[E + e];
    float norm = dis[src] * dis[dst];
    v4bf tv = *(const v4bf*)(t + (size_t)src * FEAT + q * 4);
    float* hp = h + (size_t)dst * FEAT + q * 4;
#pragma unroll
    for (int k = 0; k < 4; ++k)
        atomicAdd(hp + k, (float)tv[k] * norm);
}

// ---------------- pooling ----------------

__global__ void pool_cnt(const int* __restrict__ batch, float* __restrict__ cnt, int n) {
    int i = blockIdx.x * blockDim.x + threadIdx.x;
    if (i < n) atomicAdd(&cnt[batch[i]], 1.0f);
}

__global__ void pool_sum(const int* __restrict__ batch, const float* __restrict__ h,
                         float* __restrict__ sums, int n128)
{
    int gid = blockIdx.x * blockDim.x + threadIdx.x;
    if (gid >= n128) return;
    int i = gid >> 7;
    int f = gid & 127;
    atomicAdd(&sums[batch[i] * FEAT + f], fmaxf(h[gid], 0.0f));  // fused ReLU
}

__global__ void pool_fin(const float* __restrict__ sums, const float* __restrict__ cnt,
                         __bf16* __restrict__ zb, int m128)
{
    int gid = blockIdx.x * blockDim.x + threadIdx.x;
    if (gid >= m128) return;
    int m = gid >> 7;
    zb[gid] = (__bf16)(sums[gid] / fmaxf(cnt[m], 1.0f));
}

// ---------------- host ----------------

extern "C" void kernel_launch(void* const* d_in, const int* in_sizes, int n_in,
                              void* d_out, int out_size, void* d_ws, size_t ws_size,
                              hipStream_t stream)
{
    const float* x   = (const float*)d_in[0];
    const int*   ei  = (const int*)d_in[1];
    const int*   bat = (const int*)d_in[2];
    const float* W1  = (const float*)d_in[3];
    const float* b1  = (const float*)d_in[4];
    const float* W2  = (const float*)d_in[5];
    const float* b2  = (const float*)d_in[6];
    const float* Wf  = (const float*)d_in[7];
    const float* bf  = (const float*)d_in[8];
    float*       out = (float*)d_out;

    const int N = in_sizes[0] / FEAT;
    const int E = in_sizes[1] / 2;
    const int M = out_size / FEAT;
    const size_t NF = (size_t)N * FEAT;
    const size_t MF = (size_t)M * FEAT;

    // workspace carve-out (256B aligned)
    char* base = (char*)d_ws;
    size_t off = 0;
    auto carve = [&](size_t bytes) -> char* {
        char* p = base + off;
        off = (off + bytes + 255) & ~(size_t)255;
        return p;
    };
    float*  dis  = (float*) carve((size_t)N * 4);          // degree, then rsqrt in place
    __bf16* xb   = (__bf16*)carve(NF * 2);                 // bf16 layer input
    __bf16* tb   = (__bf16*)carve(NF * 2);                 // bf16 GEMM output (pre-aggregation)
    float*  hf   = (float*) carve(NF * 4);                 // f32 aggregation target
    __bf16* w1s  = (__bf16*)carve(16384 * 2);
    __bf16* w2s  = (__bf16*)carve(16384 * 2);
    __bf16* wfs  = (__bf16*)carve(16384 * 2);
    float*  sums = (float*) carve(MF * 4);
    float*  cnt  = (float*) carve((size_t)M * 4);
    __bf16* zb   = (__bf16*)carve(MF * 2);
    (void)ws_size; (void)n_in;

    const int TPB = 256;
    auto blocks = [](size_t n, int tpb) { return (unsigned)((n + tpb - 1) / tpb); };

    // 1) symmetric normalization: deg (init 1.0 for self-loop) -> dis = rsqrt(deg)
    fill_f32<<<blocks(N, TPB), TPB, 0, stream>>>(dis, 1.0f, N);
    deg_edges<<<blocks(E, TPB), TPB, 0, stream>>>(ei, dis, E);
    rsqrt_k<<<blocks(N, TPB), TPB, 0, stream>>>(dis, N);

    // 2) weights -> swizzled bf16; x -> bf16
    swizzle_w<<<blocks(16384, TPB), TPB, 0, stream>>>(W1, w1s);
    swizzle_w<<<blocks(16384, TPB), TPB, 0, stream>>>(W2, w2s);
    swizzle_w<<<blocks(16384, TPB), TPB, 0, stream>>>(Wf, wfs);
    f32_to_bf16<<<blocks(NF, TPB), TPB, 0, stream>>>(x, xb, (int)NF);

    const unsigned gemmN = blocks(((N + 15) / 16), 8);   // 8 waves/block, 1 strip/wave
    const unsigned gemmM = blocks(((M + 15) / 16), 8);

    // 3) layer 1: t = x@W1 ; h = aggregate(t) + b1
    gemm128_wmma<<<gemmN, TPB, 0, stream>>>(xb, w1s, nullptr, nullptr, tb, N);
    agg_init<<<blocks(NF, TPB), TPB, 0, stream>>>(tb, dis, b1, hf, (int)NF);
    agg_edges<<<blocks((size_t)E * 32, TPB), TPB, 0, stream>>>(ei, tb, dis, hf, E);
    relu_to_bf16<<<blocks(NF, TPB), TPB, 0, stream>>>(hf, xb, (int)NF);

    // 4) layer 2
    gemm128_wmma<<<gemmN, TPB, 0, stream>>>(xb, w2s, nullptr, nullptr, tb, N);
    agg_init<<<blocks(NF, TPB), TPB, 0, stream>>>(tb, dis, b2, hf, (int)NF);
    agg_edges<<<blocks((size_t)E * 32, TPB), TPB, 0, stream>>>(ei, tb, dis, hf, E);

    // 5) global mean pool (ReLU fused into the sum)
    fill_f32<<<blocks(MF, TPB), TPB, 0, stream>>>(sums, 0.0f, (int)MF);
    fill_f32<<<blocks(M, TPB), TPB, 0, stream>>>(cnt, 0.0f, M);
    pool_cnt<<<blocks(N, TPB), TPB, 0, stream>>>(bat, cnt, N);
    pool_sum<<<blocks(NF, TPB), TPB, 0, stream>>>(bat, hf, sums, (int)NF);
    pool_fin<<<blocks(MF, TPB), TPB, 0, stream>>>(sums, cnt, zb, (int)MF);

    // 6) head: out = z @ Wf + bf  (bias fused, f32 output)
    gemm128_wmma<<<gemmM, TPB, 0, stream>>>(zb, wfs, bf, out, nullptr, M);
}
